// MultiHeadSelfAttention_72224170049917
// MI455X (gfx1250) — matrix operand order repaired and checked by reading, not compiled, not run on hardware
//
#include <hip/hip_runtime.h>
#include <hip/hip_bf16.h>
#include <math.h>

#define Bn 4
#define Sn 2048
#define Dn 1024
#define Hn 16
#define En 64
#define ROWS (Bn * Sn) /* 8192 */

typedef __attribute__((ext_vector_type(16))) __bf16 v16bf;
typedef __attribute__((ext_vector_type(8)))  __bf16 v8bf;
typedef __attribute__((ext_vector_type(8)))  float  v8f;

static __device__ __forceinline__ v8f wmma_bf16(v16bf a, v16bf b, v8f c) {
  // D(f32 16x16) = A(bf16 16x32) * B(bf16 32x16) + C
  return __builtin_amdgcn_wmma_f32_16x16x32_bf16(false, a, false, b, (short)0, c,
                                                 false, false);
}

static __device__ __forceinline__ v16bf mk16(v8bf lo, v8bf hi) {
  v16bf r;
#pragma unroll
  for (int i = 0; i < 8; ++i) { r[i] = lo[i]; r[8 + i] = hi[i]; }
  return r;
}

static __device__ __forceinline__ v8bf ld8(const __bf16* p) {
  return *(const v8bf*)p;
}

static __device__ __forceinline__ v8f v8f_zero() {
  v8f z;
#pragma unroll
  for (int i = 0; i < 8; ++i) z[i] = 0.0f;
  return z;
}

// ---------------------------------------------------------------- prep kernels
__global__ void cvt_kernel(const float* __restrict__ src, __bf16* __restrict__ dst,
                           int n) {
  int i = blockIdx.x * blockDim.x + threadIdx.x;
  int stride = gridDim.x * blockDim.x;
  for (; i < n; i += stride) dst[i] = (__bf16)src[i];
}

// W[h][d][e] (f32) -> Wt[h][e][d] (bf16) so GEMM B-operand reads are contiguous in K
__global__ void wtr_kernel(const float* __restrict__ w, __bf16* __restrict__ wt) {
  int i = blockIdx.x * blockDim.x + threadIdx.x; // exactly H*D*E = 1M threads
  int h = i / (Dn * En);
  int rem = i - h * (Dn * En);
  int d = rem / En;
  int e = rem - d * En;
  wt[(h * En + e) * Dn + d] = (__bf16)w[i];
}

// ---------------------------------------------------------------- QKV projection
// One wave computes a 32x32 block (2x2 WMMA tiles) of Q, K, or V for one head.
// K-loop: 32 iters x 4 wmma_bf16; A reused across column tiles, B across row tiles.
__global__ __launch_bounds__(256) void qkv_kernel(
    const __bf16* __restrict__ X,   // [8192,1024] bf16
    const __bf16* __restrict__ Wqt, // [H,E,D]
    const __bf16* __restrict__ Wkt,
    const __bf16* __restrict__ Wvt,
    __bf16* __restrict__ Q,  // [B,H,S,E], pre-scaled by 1/sqrt(E)
    __bf16* __restrict__ K,  // [B,H,S,E]
    __bf16* __restrict__ Vt) // [B,H,E,S] (transposed for PV GEMM)
{
  int w = (int)((blockIdx.x * blockDim.x + threadIdx.x) >> 5);
  int lane = threadIdx.x & 31;
  int half = lane >> 4, l16 = lane & 15;

  int ct = w & 1;        w >>= 1;   // 32-col tile (E split into 2)
  int rt = w & 255;      w >>= 8;   // 32-row tile (8192/32)
  int h  = w & 15;       w >>= 4;
  int mat = w;                      // 0=Q 1=K 2=V
  if (mat >= 3) return;

  const __bf16* Wt = (mat == 0) ? Wqt : (mat == 1) ? Wkt : Wvt;

  const __bf16* Ar0 = X + (size_t)(rt * 32 + l16) * Dn + half * 8;
  const __bf16* Ar1 = Ar0 + (size_t)16 * Dn;
  const __bf16* Br0 =
      Wt + (size_t)h * En * Dn + (size_t)(ct * 32 + l16) * Dn + half * 8;
  const __bf16* Br1 = Br0 + (size_t)16 * Dn;

  v8f acc00 = v8f_zero(), acc01 = v8f_zero();
  v8f acc10 = v8f_zero(), acc11 = v8f_zero();

#pragma unroll 2
  for (int k0 = 0; k0 < Dn; k0 += 32) {
    __builtin_prefetch(Ar0 + k0 + 128, 0, 3);
    __builtin_prefetch(Ar1 + k0 + 128, 0, 3);
    __builtin_prefetch(Br0 + k0 + 128, 0, 3);
    __builtin_prefetch(Br1 + k0 + 128, 0, 3);
    v16bf a0 = mk16(ld8(Ar0 + k0), ld8(Ar0 + k0 + 16));
    v16bf a1 = mk16(ld8(Ar1 + k0), ld8(Ar1 + k0 + 16));
    v16bf b0 = mk16(ld8(Br0 + k0), ld8(Br0 + k0 + 16));
    v16bf b1 = mk16(ld8(Br1 + k0), ld8(Br1 + k0 + 16));
    acc00 = wmma_bf16(a0, b0, acc00);
    acc01 = wmma_bf16(a0, b1, acc01);
    acc10 = wmma_bf16(a1, b0, acc10);
    acc11 = wmma_bf16(a1, b1, acc11);
  }

  int row_blk = rt * 32;
  int b = row_blk / Sn;               // whole 32-row block in one batch (Sn%32==0)
  int s_base = row_blk - b * Sn + half * 8;
  int col_base = ct * 32 + l16;
  v8f* accs[2][2] = {{&acc00, &acc01}, {&acc10, &acc11}};

  if (mat == 0) { // Q, scaled by 1/sqrt(64)
    __bf16* base = Q + ((size_t)(b * Hn + h) * Sn) * En;
#pragma unroll
    for (int i = 0; i < 2; ++i)
#pragma unroll
      for (int j = 0; j < 2; ++j) {
        int col = col_base + j * 16;
#pragma unroll
        for (int r = 0; r < 8; ++r)
          base[(size_t)(s_base + i * 16 + r) * En + col] =
              (__bf16)((*accs[i][j])[r] * 0.125f);
      }
  } else if (mat == 1) { // K
    __bf16* base = K + ((size_t)(b * Hn + h) * Sn) * En;
#pragma unroll
    for (int i = 0; i < 2; ++i)
#pragma unroll
      for (int j = 0; j < 2; ++j) {
        int col = col_base + j * 16;
#pragma unroll
        for (int r = 0; r < 8; ++r)
          base[(size_t)(s_base + i * 16 + r) * En + col] =
              (__bf16)(*accs[i][j])[r];
      }
  } else { // V transposed: Vt[(bh*En + e)*Sn + s]
    __bf16* base = Vt + ((size_t)(b * Hn + h) * En) * Sn;
#pragma unroll
    for (int i = 0; i < 2; ++i)
#pragma unroll
      for (int j = 0; j < 2; ++j) {
        int col = col_base + j * 16;
#pragma unroll
        for (int r = 0; r < 8; ++r)
          base[(size_t)col * Sn + (s_base + i * 16 + r)] =
              (__bf16)(*accs[i][j])[r];
      }
  }
}

// ---------------------------------------------------------------- flash attention
// One wave per (b,h, 16-row q-tile). 32-key chunks, online softmax, causal.
__global__ __launch_bounds__(128) void attn_kernel(
    const __bf16* __restrict__ Q,  // [B,H,S,E] (pre-scaled)
    const __bf16* __restrict__ K,  // [B,H,S,E]
    const __bf16* __restrict__ Vt, // [B,H,E,S]
    __bf16* __restrict__ AO)       // [B,S,D] bf16 (heads concatenated)
{
  __shared__ __align__(16) __bf16 lds_buf[4 * 16 * 32]; // 1KB per wave
  int wave = threadIdx.x >> 5;
  int lane = threadIdx.x & 31;
  int half = lane >> 4, l16 = lane & 15;

  int w = blockIdx.x * 4 + wave;
  int qt = w & 127;
  int bh = w >> 7; // 0..63
  int q0 = qt * 16;

  const __bf16* Qb = Q + (size_t)bh * Sn * En;
  const __bf16* Kb = K + (size_t)bh * Sn * En;
  const __bf16* Vb = Vt + (size_t)bh * En * Sn;
  __bf16* lw = lds_buf + wave * (16 * 32);

  // Q A-operands for the two K-dim (E) steps, resident for the whole tile
  const __bf16* qp = Qb + (size_t)(q0 + l16) * En + half * 8;
  v16bf qa0 = mk16(ld8(qp), ld8(qp + 16));
  v16bf qa1 = mk16(ld8(qp + 32), ld8(qp + 48));

  v8f acc[4];
  float m_run[8], l_run[8];
#pragma unroll
  for (int et = 0; et < 4; ++et) acc[et] = v8f_zero();
#pragma unroll
  for (int r = 0; r < 8; ++r) { m_run[r] = -__builtin_inff(); l_run[r] = 0.0f; }

  int nch = q0 / 32 + 1;
  for (int ch = 0; ch < nch; ++ch) {
    int t0 = ch * 32;

    // scores: S[16q x 32kv] via 4 WMMAs (2 col-groups x 2 E-steps)
    const __bf16* kp0 = Kb + (size_t)(t0 + l16) * En + half * 8;
    const __bf16* kp1 = kp0 + (size_t)16 * En;
    __builtin_prefetch(kp0 + (size_t)32 * En, 0, 3); // next chunk's K
    v8f s0 = v8f_zero(), s1 = v8f_zero();
    s0 = wmma_bf16(qa0, mk16(ld8(kp0), ld8(kp0 + 16)), s0);
    s0 = wmma_bf16(qa1, mk16(ld8(kp0 + 32), ld8(kp0 + 48)), s0);
    s1 = wmma_bf16(qa0, mk16(ld8(kp1), ld8(kp1 + 16)), s1);
    s1 = wmma_bf16(qa1, mk16(ld8(kp1 + 32), ld8(kp1 + 48)), s1);

    // causal mask (only the diagonal chunk needs it)
    int rbase = q0 + half * 8;
    int col0 = t0 + l16, col1 = col0 + 16;
    if (t0 + 31 > q0) {
#pragma unroll
      for (int r = 0; r < 8; ++r) {
        if (col0 > rbase + r) s0[r] = -__builtin_inff();
        if (col1 > rbase + r) s1[r] = -__builtin_inff();
      }
    }

    // row-max of the chunk (reduce across each 16-lane half)
    float mc[8];
#pragma unroll
    for (int r = 0; r < 8; ++r) mc[r] = fmaxf(s0[r], s1[r]);
#pragma unroll
    for (int off = 1; off < 16; off <<= 1) {
#pragma unroll
      for (int r = 0; r < 8; ++r) mc[r] = fmaxf(mc[r], __shfl_xor(mc[r], off, 32));
    }

    float p0[8], p1[8], al[8], ls[8];
#pragma unroll
    for (int r = 0; r < 8; ++r) {
      float mnew = fmaxf(m_run[r], mc[r]);
      al[r] = __expf(m_run[r] - mnew);
      p0[r] = __expf(s0[r] - mnew);
      p1[r] = __expf(s1[r] - mnew);
      m_run[r] = mnew;
      ls[r] = p0[r] + p1[r];
    }
#pragma unroll
    for (int off = 1; off < 16; off <<= 1) {
#pragma unroll
      for (int r = 0; r < 8; ++r) ls[r] += __shfl_xor(ls[r], off, 32);
    }
#pragma unroll
    for (int r = 0; r < 8; ++r) l_run[r] = l_run[r] * al[r] + ls[r];
#pragma unroll
    for (int et = 0; et < 4; ++et)
#pragma unroll
      for (int r = 0; r < 8; ++r) acc[et][r] *= al[r];

    // P: D-layout -> A-layout via LDS (per-wave private region; DS is in-order)
#pragma unroll
    for (int r = 0; r < 8; ++r) {
      int row = half * 8 + r;
      lw[row * 32 + l16]      = (__bf16)p0[r];
      lw[row * 32 + 16 + l16] = (__bf16)p1[r];
    }
    asm volatile("s_wait_dscnt 0x0" ::: "memory");
    const __bf16* pp = lw + l16 * 32 + half * 8;
    v16bf pa = mk16(*(const v8bf*)pp, *(const v8bf*)(pp + 16));

    // O[16q x 64e] += P x V  (4 WMMAs, B from transposed V: contiguous in kv)
#pragma unroll
    for (int et = 0; et < 4; ++et) {
      const __bf16* vp = Vb + (size_t)(et * 16 + l16) * Sn + t0 + half * 8;
      __builtin_prefetch(vp + 32, 0, 3); // next chunk's V
      acc[et] = wmma_bf16(pa, mk16(ld8(vp), ld8(vp + 16)), acc[et]);
    }
  }

  // normalize and write concat-head output (bf16) for the final projection GEMM
  int bb = bh >> 4, hh = bh & 15;
#pragma unroll
  for (int r = 0; r < 8; ++r) {
    float inv = 1.0f / l_run[r];
    int s = q0 + half * 8 + r;
    size_t base = ((size_t)bb * Sn + s) * Dn + hh * En + l16;
#pragma unroll
    for (int et = 0; et < 4; ++et)
      AO[base + et * 16] = (__bf16)(acc[et][r] * inv);
  }
}

// ---------------------------------------------------------------- output projection
// out[s][n] = sum_k AO[s][k] * Wo[n][k] + bo[n]  (Wo row-major == Bt layout)
// One wave computes a 32x32 block (2x2 WMMA tiles).
__global__ __launch_bounds__(256) void proj_kernel(
    const __bf16* __restrict__ AO,  // [8192,1024]
    const __bf16* __restrict__ Wob, // [1024,1024] bf16
    const float* __restrict__ bo,   // [1024]
    float* __restrict__ out)        // [8192,1024] f32
{
  int w = (int)((blockIdx.x * blockDim.x + threadIdx.x) >> 5);
  int lane = threadIdx.x & 31;
  int half = lane >> 4, l16 = lane & 15;

  int nt = w & 31;  // 32-col tile (1024/32)
  int rt = w >> 5;  // 32-row tile (8192/32), 0..255

  const __bf16* Ar0 = AO + (size_t)(rt * 32 + l16) * Dn + half * 8;
  const __bf16* Ar1 = Ar0 + (size_t)16 * Dn;
  const __bf16* Br0 = Wob + (size_t)(nt * 32 + l16) * Dn + half * 8;
  const __bf16* Br1 = Br0 + (size_t)16 * Dn;

  v8f acc00 = v8f_zero(), acc01 = v8f_zero();
  v8f acc10 = v8f_zero(), acc11 = v8f_zero();

#pragma unroll 2
  for (int k0 = 0; k0 < Dn; k0 += 32) {
    __builtin_prefetch(Ar0 + k0 + 128, 0, 3);
    __builtin_prefetch(Ar1 + k0 + 128, 0, 3);
    __builtin_prefetch(Br0 + k0 + 128, 0, 3);
    __builtin_prefetch(Br1 + k0 + 128, 0, 3);
    v16bf a0 = mk16(ld8(Ar0 + k0), ld8(Ar0 + k0 + 16));
    v16bf a1 = mk16(ld8(Ar1 + k0), ld8(Ar1 + k0 + 16));
    v16bf b0 = mk16(ld8(Br0 + k0), ld8(Br0 + k0 + 16));
    v16bf b1 = mk16(ld8(Br1 + k0), ld8(Br1 + k0 + 16));
    acc00 = wmma_bf16(a0, b0, acc00);
    acc01 = wmma_bf16(a0, b1, acc01);
    acc10 = wmma_bf16(a1, b0, acc10);
    acc11 = wmma_bf16(a1, b1, acc11);
  }

  int row0 = rt * 32 + half * 8;
  int col0 = nt * 32 + l16;
  float bias0 = bo[col0];
  float bias1 = bo[col0 + 16];
  v8f* accs[2][2] = {{&acc00, &acc01}, {&acc10, &acc11}};
#pragma unroll
  for (int i = 0; i < 2; ++i)
#pragma unroll
    for (int r = 0; r < 8; ++r) {
      size_t rowoff = (size_t)(row0 + i * 16 + r) * Dn;
      out[rowoff + col0]      = (*accs[i][0])[r] + bias0;
      out[rowoff + col0 + 16] = (*accs[i][1])[r] + bias1;
    }
}

// ---------------------------------------------------------------- launcher
extern "C" void kernel_launch(void* const* d_in, const int* in_sizes, int n_in,
                              void* d_out, int out_size, void* d_ws, size_t ws_size,
                              hipStream_t stream) {
  const float* emb = (const float*)d_in[0]; // [B,S,D]
  const float* Wq  = (const float*)d_in[1]; // [H,D,E]
  const float* Wk  = (const float*)d_in[2];
  const float* Wv  = (const float*)d_in[3];
  const float* Wo  = (const float*)d_in[4]; // [D,D]
  const float* bo  = (const float*)d_in[5]; // [D]
  float* out = (float*)d_out;

  __bf16* ws = (__bf16*)d_ws;
  size_t off = 0;
  __bf16* Xb  = ws + off; off += (size_t)ROWS * Dn;      // 8M
  __bf16* Wqt = ws + off; off += (size_t)Hn * En * Dn;   // 1M
  __bf16* Wkt = ws + off; off += (size_t)Hn * En * Dn;
  __bf16* Wvt = ws + off; off += (size_t)Hn * En * Dn;
  __bf16* Wob = ws + off; off += (size_t)Dn * Dn;        // 1M
  __bf16* Qb  = ws + off; off += (size_t)Bn * Hn * Sn * En; // 8M
  __bf16* Kb  = ws + off; off += (size_t)Bn * Hn * Sn * En;
  __bf16* Vtb = ws + off; off += (size_t)Bn * Hn * En * Sn;
  __bf16* AOb = ws + off; off += (size_t)ROWS * Dn;

  cvt_kernel<<<4096, 256, 0, stream>>>(emb, Xb, ROWS * Dn);
  cvt_kernel<<<1024, 256, 0, stream>>>(Wo, Wob, Dn * Dn);
  wtr_kernel<<<4096, 256, 0, stream>>>(Wq, Wqt);
  wtr_kernel<<<4096, 256, 0, stream>>>(Wk, Wkt);
  wtr_kernel<<<4096, 256, 0, stream>>>(Wv, Wvt);

  // 3*16*256*2 = 24576 block-waves, 8 waves/block
  qkv_kernel<<<3072, 256, 0, stream>>>(Xb, Wqt, Wkt, Wvt, Qb, Kb, Vtb);
  // 64*128 = 8192 waves, 4 waves/block
  attn_kernel<<<2048, 128, 0, stream>>>(Qb, Kb, Vtb, AOb);
  // 256*32 = 8192 block-waves, 8 waves/block
  proj_kernel<<<1024, 256, 0, stream>>>(AOb, Wob, bo, out);
}
